// MultiHeadAttention_3633542333276
// MI455X (gfx1250) — matrix operand order repaired and checked by reading, not compiled, not run on hardware
//
#include <hip/hip_runtime.h>
#include <cstdint>

// ---------- types ----------
typedef __bf16 bf16_t;
typedef __attribute__((ext_vector_type(16))) __bf16 v16bf;
typedef __attribute__((ext_vector_type(8)))  __bf16 v8bf;
typedef __attribute__((ext_vector_type(8)))  float  v8f;

#define HEADS 16
#define KV_HEADS 4
#define HEAD_DIM 64
#define HIDDEN 1024
#define SEQ 2048
#define BATCH 2
#define TOKENS (BATCH * SEQ)   // 4096

static __device__ inline v16bf combine16(v8bf lo, v8bf hi) {
  return __builtin_shufflevector(lo, hi, 0,1,2,3,4,5,6,7,8,9,10,11,12,13,14,15);
}

static __device__ inline v8f wmma_bf16(v16bf a, v16bf b, v8f c) {
  // D = A(16x32 bf16) * B(32x16 bf16) + C(16x16 f32)
  return __builtin_amdgcn_wmma_f32_16x16x32_bf16(false, a, false, b, (short)0, c, false, false);
}

// ---- CDNA5 async global->LDS copy (16B per lane), tracked by ASYNCcnt ----
// GVS form: mem_addr = SGPR_base + VGPR_offset; VDST VGPR carries LDS byte addr.
// Generic->LDS address: low 32 bits of the generic pointer (aperture rule, ISA 10.2).
static __device__ inline void async_load_b128(void* lds_ptr, uint32_t glob_off,
                                              const void* glob_base) {
  uint32_t lds_addr = (uint32_t)(uintptr_t)lds_ptr;
  asm volatile("global_load_async_to_lds_b128 %0, %1, %2"
               :: "v"(lds_addr), "v"(glob_off), "s"(glob_base)
               : "memory");
}
static __device__ inline void wait_async0() {
  asm volatile("s_wait_asynccnt 0x0" ::: "memory");
}

// ---------- fp32 -> bf16 ----------
__global__ void cvt_f32_bf16(const float* __restrict__ in, bf16_t* __restrict__ out, int n) {
  int i = blockIdx.x * blockDim.x + threadIdx.x;
  if (i < n) out[i] = (bf16_t)in[i];
}

// ---------- RoPE (in-place on bf16 [token, head*64+d]) ----------
__global__ void rope_kernel(bf16_t* __restrict__ X, int heads, int total) {
  int idx = blockIdx.x * blockDim.x + threadIdx.x;
  if (idx >= total) return;
  int i    = idx & 31;          // 0..31 : rotation pair index
  int rest = idx >> 5;
  int h    = rest % heads;
  int tok  = rest / heads;
  int s    = tok & (SEQ - 1);
  size_t base = ((size_t)tok * heads + h) * HEAD_DIM;
  // inv_freq[i] = 10000^(-i/32) = exp(-i * ln(10000)/32)
  float invf = __expf(-(float)i * 0.28782313662425572f);
  float ang  = (float)s * invf;
  float c, sn; __sincosf(ang, &c, &sn);
  float x1 = (float)X[base + i];
  float x2 = (float)X[base + i + 32];
  X[base + i]      = (bf16_t)(x1 * c - x2 * sn);
  X[base + i + 32] = (bf16_t)(x2 * c + x1 * sn);
}

// ---------- tiled bf16 GEMM: C[M,N] = A[M,K] @ B[K,N] ----------
// 256 threads = 8 waves (wave32). Block tile 128x128, K-step 32.
// Wave grid 4(M) x 2(N): each wave computes 32x64 via 2x4 WMMA accumulators.
// A tile staged with async global->LDS (no transpose needed); B tile staged
// via coalesced loads + LDS transpose-scatter (fragment wants [n][k]).
#define BM 128
#define BN 128
#define BK 32

template <bool OUT_F32>
__global__ void __launch_bounds__(256)
gemm_bf16_kernel(const bf16_t* __restrict__ A, const bf16_t* __restrict__ B,
                 void* __restrict__ Cout, int M, int N, int K) {
  __shared__ __align__(32) bf16_t As[BM][BK];  // row-major tile        (8 KB)
  __shared__ __align__(32) bf16_t Bt[BN][BK];  // transposed [n][k]     (8 KB)

  const int tid  = threadIdx.x;
  const int lane = tid & 31;
  const int wave = tid >> 5;
  const int wm   = wave & 3;    // 0..3 -> 32-row strip
  const int wn   = wave >> 2;   // 0..1 -> 64-col strip
  const int m0   = blockIdx.y * BM;
  const int n0   = blockIdx.x * BN;
  const int half = lane >> 4;   // lane group
  const int l15  = lane & 15;
  const int kA   = half * 16;   // A-fragment k_base
  const int kB   = half * 16;   // B-fragment k_base

  v8f acc[2][4];
#pragma unroll
  for (int mf = 0; mf < 2; ++mf)
#pragma unroll
    for (int nf = 0; nf < 4; ++nf)
#pragma unroll
      for (int e = 0; e < 8; ++e) acc[mf][nf][e] = 0.f;

  const int kTiles = K / BK;
  for (int kt = 0; kt < kTiles; ++kt) {
    const int k0 = kt * BK;
    __syncthreads();
    // stage A: 128x32 bf16 = 512 chunks of 16B, async DMA into LDS
#pragma unroll
    for (int c = tid; c < 512; c += 256) {
      int r = c >> 2, co = (c & 3) * 8;
      uint32_t goff = (uint32_t)(((m0 + r) * K + k0 + co) * (int)sizeof(bf16_t));
      async_load_b128(&As[r][co], goff, A);
    }
    // stage B: read 32x128 coalesced, transpose-scatter into Bt[n][k]
#pragma unroll
    for (int c = tid; c < 512; c += 256) {
      int kr = c >> 4, nc = (c & 15) * 8;
      v8bf t = *(const v8bf*)&B[(size_t)(k0 + kr) * N + n0 + nc];
#pragma unroll
      for (int e = 0; e < 8; ++e) Bt[nc + e][kr] = t[e];
    }
    wait_async0();
    __syncthreads();

    v16bf af[2], bfr[4];
#pragma unroll
    for (int mf = 0; mf < 2; ++mf) {
      const bf16_t* rp = &As[wm * 32 + mf * 16 + l15][0];
      af[mf] = combine16(*(const v8bf*)(rp + kA), *(const v8bf*)(rp + 8 + kA));
    }
#pragma unroll
    for (int nf = 0; nf < 4; ++nf)
      bfr[nf] = *(const v16bf*)&Bt[wn * 64 + nf * 16 + l15][kB];

#pragma unroll
    for (int mf = 0; mf < 2; ++mf)
#pragma unroll
      for (int nf = 0; nf < 4; ++nf)
        acc[mf][nf] = wmma_bf16(af[mf], bfr[nf], acc[mf][nf]);
  }

  // epilogue: D-fragment row = i + 8*half, col = l15
#pragma unroll
  for (int mf = 0; mf < 2; ++mf)
#pragma unroll
    for (int nf = 0; nf < 4; ++nf)
#pragma unroll
      for (int i = 0; i < 8; ++i) {
        size_t row = (size_t)(m0 + wm * 32 + mf * 16 + half * 8 + i);
        size_t col = (size_t)(n0 + wn * 64 + nf * 16 + l15);
        if (OUT_F32)
          ((float*)Cout)[row * N + col] = acc[mf][nf][i];
        else
          ((bf16_t*)Cout)[row * N + col] = (bf16_t)acc[mf][nf][i];
      }
}

// ---------- flash attention (causal, GQA) ----------
// Block = 128 threads = 4 waves; one 64-row query block of one (b,h).
// Wave w owns 16 query rows. Key loop in 64-row blocks up to the diagonal.
// K tile goes async global->LDS (row-major K *is* the B-operand layout for
// Q@K^T); V tile needs a transpose so it uses load + LDS scatter.
__global__ void __launch_bounds__(128)
attn_kernel(const bf16_t* __restrict__ Q, const bf16_t* __restrict__ Kg,
            const bf16_t* __restrict__ Vg, bf16_t* __restrict__ AO) {
  __shared__ __align__(32) bf16_t Ks[64][64];      // K rows, row-major   (8 KB)
  __shared__ __align__(32) bf16_t Vt[64][64];      // V transposed [d][k] (8 KB)
  __shared__ __align__(32) bf16_t Ps[4][16][64];   // per-wave P tiles    (8 KB)

  const int tid  = threadIdx.x;
  const int lane = tid & 31;
  const int wave = tid >> 5;
  const int qb   = blockIdx.x;        // 0..31 (query block of 64)
  const int h    = blockIdx.y;        // 0..15
  const int b    = blockIdx.z;        // 0..1
  const int hk   = h >> 2;            // kv head (G = 4)
  const int half = lane >> 4;
  const int l15  = lane & 15;
  const int kA   = half * 16;
  const int kB   = half * 16;

  // ---- load this wave's Q A-fragments (16 rows x 64 d), kept in VGPRs ----
  v16bf aq[2];
  {
    size_t qtok = (size_t)(b * SEQ + qb * 64 + wave * 16 + l15);
    const bf16_t* qp = Q + qtok * (HEADS * HEAD_DIM) + h * HEAD_DIM;
#pragma unroll
    for (int kk = 0; kk < 2; ++kk)
      aq[kk] = combine16(*(const v8bf*)(qp + kk * 32 + kA),
                         *(const v8bf*)(qp + kk * 32 + 8 + kA));
  }

  v8f o[4];
  float mrow[8], lrow[8];
#pragma unroll
  for (int nf = 0; nf < 4; ++nf)
#pragma unroll
    for (int e = 0; e < 8; ++e) o[nf][e] = 0.f;
#pragma unroll
  for (int i = 0; i < 8; ++i) { mrow[i] = -3.0e38f; lrow[i] = 0.f; }

  const int rowBase = qb * 64 + wave * 16 + half * 8;  // global q row of elem i

  const int nkb = qb + 1;   // causal: key blocks 0..qb
  for (int kbk = 0; kbk < nkb; ++kbk) {
    __syncthreads();
    // stage K (async DMA, row-major) and V (transposed via scatter), head hk
#pragma unroll
    for (int c = tid; c < 512; c += 128) {
      int kr = c >> 3, d0 = (c & 7) * 8;
      int ktok = b * SEQ + kbk * 64 + kr;
      uint32_t goff =
          (uint32_t)(((ktok * KV_HEADS + hk) * HEAD_DIM + d0) * (int)sizeof(bf16_t));
      async_load_b128(&Ks[kr][d0], goff, Kg);
      v8bf vv = *(const v8bf*)(Vg + ((size_t)ktok * KV_HEADS + hk) * HEAD_DIM + d0);
#pragma unroll
      for (int e = 0; e < 8; ++e) Vt[d0 + e][kr] = vv[e];
    }
    wait_async0();
    __syncthreads();

    // ---- scores S = Q @ K^T : 16q x 64k per wave ----
    v8f sc[4];
#pragma unroll
    for (int nf = 0; nf < 4; ++nf)
#pragma unroll
      for (int e = 0; e < 8; ++e) sc[nf][e] = 0.f;
#pragma unroll
    for (int kk = 0; kk < 2; ++kk)
#pragma unroll
      for (int nf = 0; nf < 4; ++nf) {
        v16bf bk = *(const v16bf*)&Ks[nf * 16 + l15][kk * 32 + kB];
        sc[nf] = wmma_bf16(aq[kk], bk, sc[nf]);
      }

    // scale + causal mask
#pragma unroll
    for (int nf = 0; nf < 4; ++nf) {
      int col = kbk * 64 + nf * 16 + l15;
#pragma unroll
      for (int i = 0; i < 8; ++i) {
        float s = sc[nf][i] * 0.125f;           // 1/sqrt(64)
        if (col > rowBase + i) s = -1.0e9f;     // additive causal mask
        sc[nf][i] = s;
      }
    }

    // ---- online softmax per row (16-lane group reductions) ----
#pragma unroll
    for (int i = 0; i < 8; ++i) {
      float mx = fmaxf(fmaxf(sc[0][i], sc[1][i]), fmaxf(sc[2][i], sc[3][i]));
#pragma unroll
      for (int off = 8; off >= 1; off >>= 1) mx = fmaxf(mx, __shfl_xor(mx, off, 32));
      float mnew  = fmaxf(mrow[i], mx);
      float alpha = __expf(mrow[i] - mnew);
      mrow[i] = mnew;
      float ps = 0.f;
#pragma unroll
      for (int nf = 0; nf < 4; ++nf) {
        float p = __expf(sc[nf][i] - mnew);
        sc[nf][i] = p;
        ps += p;
      }
#pragma unroll
      for (int off = 8; off >= 1; off >>= 1) ps += __shfl_xor(ps, off, 32);
      lrow[i] = lrow[i] * alpha + ps;
#pragma unroll
      for (int nf = 0; nf < 4; ++nf) o[nf][i] *= alpha;
    }

    // ---- write P (C-layout) to per-wave LDS, re-read as A-fragments ----
#pragma unroll
    for (int nf = 0; nf < 4; ++nf)
#pragma unroll
      for (int i = 0; i < 8; ++i)
        Ps[wave][half * 8 + i][nf * 16 + l15] = (bf16_t)sc[nf][i];
    // same-wave DS ops are in-order on CDNA5 -> safe to read back

    // ---- O += P @ V ----
#pragma unroll
    for (int kk = 0; kk < 2; ++kk) {
      const bf16_t* pr = &Ps[wave][l15][0];
      v16bf ap = combine16(*(const v8bf*)(pr + kk * 32 + kA),
                           *(const v8bf*)(pr + kk * 32 + 8 + kA));
#pragma unroll
      for (int nf = 0; nf < 4; ++nf) {
        v16bf bv = *(const v16bf*)&Vt[nf * 16 + l15][kk * 32 + kB];
        o[nf] = wmma_bf16(ap, bv, o[nf]);
      }
    }
  }

  // ---- normalize and store to AO [token, h*64 + d] (bf16) ----
#pragma unroll
  for (int i = 0; i < 8; ++i) {
    float inv = 1.f / lrow[i];
    size_t tok = (size_t)(b * SEQ + rowBase + i);
    bf16_t* op = AO + tok * (HEADS * HEAD_DIM) + h * HEAD_DIM;
#pragma unroll
    for (int nf = 0; nf < 4; ++nf)
      op[nf * 16 + l15] = (bf16_t)(o[nf][i] * inv);
  }
}

// ---------- host launch ----------
extern "C" void kernel_launch(void* const* d_in, const int* in_sizes, int n_in,
                              void* d_out, int out_size, void* d_ws, size_t ws_size,
                              hipStream_t stream) {
  const float* X  = (const float*)d_in[0];   // [2,2048,1024]
  // d_in[1] = attention_mask (causal; applied analytically in-kernel)
  const float* Wq = (const float*)d_in[2];   // [1024,1024]
  const float* Wk = (const float*)d_in[3];   // [1024,256]
  const float* Wv = (const float*)d_in[4];   // [1024,256]
  const float* Wo = (const float*)d_in[5];   // [1024,1024]
  float* out = (float*)d_out;                // [2,2048,1024]

  const size_t nX  = (size_t)TOKENS * HIDDEN;               // 4,194,304
  const size_t nWq = (size_t)HIDDEN * HEADS * HEAD_DIM;     // 1,048,576
  const size_t nWk = (size_t)HIDDEN * KV_HEADS * HEAD_DIM;  //   262,144
  const size_t nQ  = (size_t)TOKENS * HEADS * HEAD_DIM;     // 4,194,304
  const size_t nK  = (size_t)TOKENS * KV_HEADS * HEAD_DIM;  // 1,048,576

  bf16_t* w   = (bf16_t*)d_ws;
  bf16_t* Xb  = w;              w += nX;
  bf16_t* Wqb = w;              w += nWq;
  bf16_t* Wkb = w;              w += nWk;
  bf16_t* Wvb = w;              w += nWk;
  bf16_t* Wob = w;              w += nWq;
  bf16_t* Qb  = w;              w += nQ;
  bf16_t* Kb  = w;              w += nK;
  bf16_t* Vb  = w;              w += nK;
  bf16_t* AOb = w;              w += nQ;   // ~33 MB total

  const int T = 256;
  cvt_f32_bf16<<<(int)((nX  + T - 1) / T), T, 0, stream>>>(X,  Xb,  (int)nX);
  cvt_f32_bf16<<<(int)((nWq + T - 1) / T), T, 0, stream>>>(Wq, Wqb, (int)nWq);
  cvt_f32_bf16<<<(int)((nWk + T - 1) / T), T, 0, stream>>>(Wk, Wkb, (int)nWk);
  cvt_f32_bf16<<<(int)((nWk + T - 1) / T), T, 0, stream>>>(Wv, Wvb, (int)nWk);
  cvt_f32_bf16<<<(int)((nWq + T - 1) / T), T, 0, stream>>>(Wo, Wob, (int)nWq);

  // QKV projections (bf16 out)
  gemm_bf16_kernel<false><<<dim3(HIDDEN / BN,        TOKENS / BM), 256, 0, stream>>>(
      Xb, Wqb, Qb, TOKENS, HIDDEN, HIDDEN);
  gemm_bf16_kernel<false><<<dim3((KV_HEADS*HEAD_DIM) / BN, TOKENS / BM), 256, 0, stream>>>(
      Xb, Wkb, Kb, TOKENS, KV_HEADS * HEAD_DIM, HIDDEN);
  gemm_bf16_kernel<false><<<dim3((KV_HEADS*HEAD_DIM) / BN, TOKENS / BM), 256, 0, stream>>>(
      Xb, Wvb, Vb, TOKENS, KV_HEADS * HEAD_DIM, HIDDEN);

  // RoPE on Q and K
  {
    int tq = TOKENS * HEADS * 32;
    int tk = TOKENS * KV_HEADS * 32;
    rope_kernel<<<(tq + T - 1) / T, T, 0, stream>>>(Qb, HEADS, tq);
    rope_kernel<<<(tk + T - 1) / T, T, 0, stream>>>(Kb, KV_HEADS, tk);
  }

  // flash attention: (query-block, head, batch)
  attn_kernel<<<dim3(SEQ / 64, HEADS, BATCH), 128, 0, stream>>>(Qb, Kb, Vb, AOb);

  // output projection (f32 out -> d_out)
  gemm_bf16_kernel<true><<<dim3(HIDDEN / BN, TOKENS / BM), 256, 0, stream>>>(
      AOb, Wob, out, TOKENS, HIDDEN, HIDDEN);

  (void)in_sizes; (void)n_in; (void)out_size; (void)ws_size;
}
